// S4DWrapper_41394894799307
// MI455X (gfx1250) — matrix operand order repaired and checked by reading, not compiled, not run on hardware
//
#include <hip/hip_runtime.h>
#include <hip/hip_bf16.h>
#include <math.h>

#define NB   16
#define SS   2048
#define DD   256
#define HH   512
#define NSS  64

typedef __attribute__((ext_vector_type(16))) __bf16        v16bf;
typedef __attribute__((ext_vector_type(8)))  float         v8f;
typedef __attribute__((ext_vector_type(4)))  unsigned int  u32x4;

union BfPack { u32x4 u[2]; v16bf v; };

__device__ __forceinline__ unsigned short f2bf(float f) {
    unsigned u = __builtin_bit_cast(unsigned, f);
    unsigned r = (u + 0x7FFFu + ((u >> 16) & 1u)) >> 16;
    return (unsigned short)r;
}

__device__ __forceinline__ float gelu_tanh(float x) {
    const float k0 = 0.7978845608028654f;
    const float k1 = 0.044715f;
    float t = tanhf(k0 * (x + k1 * x * x * x));
    return 0.5f * x * (1.0f + t);
}

// ---------------------------------------------------------------------------
// Kernel 1: bilinear discretization of the diagonal SSM.
// Stores per (h,n): {dA_re, dA_im, 2*Ct_re, 2*Ct_im}, Ct = C*B*dt/(1 - dtA/2)
// ---------------------------------------------------------------------------
__global__ void prep_params(const float* __restrict__ log_dt,
                            const float* __restrict__ log_A_real,
                            const float* __restrict__ A_imag,
                            const float* __restrict__ B_re, const float* __restrict__ B_im,
                            const float* __restrict__ C_re, const float* __restrict__ C_im,
                            float4* __restrict__ prm) {
    int i = blockIdx.x * blockDim.x + threadIdx.x;       // 0..H*NS-1
    int h = i >> 6;
    float dt   = expf(log_dt[h]);
    float Ar   = -expf(log_A_real[i]);
    float Ai   = A_imag[i];
    float dtAr = dt * Ar, dtAi = dt * Ai;
    float den_r = 1.0f - 0.5f * dtAr, den_i = -0.5f * dtAi;
    float num_r = 1.0f + 0.5f * dtAr, num_i =  0.5f * dtAi;
    float inv = 1.0f / (den_r * den_r + den_i * den_i);
    float dAr = (num_r * den_r + num_i * den_i) * inv;
    float dAi = (num_i * den_r - num_r * den_i) * inv;
    float cbr = C_re[i] * B_re[i] - C_im[i] * B_im[i];
    float cbi = C_re[i] * B_im[i] + C_im[i] * B_re[i];
    float tr = cbr * dt, ti = cbi * dt;
    float Ctr = (tr * den_r + ti * den_i) * inv;
    float Cti = (ti * den_r - tr * den_i) * inv;
    prm[i] = make_float4(dAr, dAi, 2.0f * Ctr, 2.0f * Cti);
}

// ---------------------------------------------------------------------------
// Kernel 2: swizzle W_glu (512x512 f32, row=k, col=n) into the WMMA B-matrix
// per-lane layout (32x16 bf16 tiles): each lane's 16 values contiguous.
// ---------------------------------------------------------------------------
__global__ void prep_w(const float* __restrict__ W, unsigned short* __restrict__ Wsw) {
    int i = blockIdx.x * 256 + threadIdx.x;              // 0..512*512-1
    int k = i >> 9, n = i & 511;
    int kb = k >> 5, kr = k & 31;
    int khalf = (kr >> 3) & 1, t = (kr & 7) | ((kr >> 4) << 3);
    int nb = n >> 4, nr = n & 15;
    int lane = khalf * 16 + nr;
    size_t dst = ((size_t)(kb * 32 + nb) * 32 + lane) * 16 + t;
    Wsw[dst] = f2bf(W[i]);
}

// ---------------------------------------------------------------------------
// Kernel 3: count_zeros per batch (for the reversal involution)
// ---------------------------------------------------------------------------
__global__ void count_zeros_k(const float* __restrict__ mask, int* __restrict__ cz) {
    int b = blockIdx.x;
    float s = 0.0f;
    for (int t = threadIdx.x; t < SS; t += 256) s += mask[b * SS + t];
    for (int off = 16; off; off >>= 1) s += __shfl_xor(s, off);
    __shared__ float sm[8];
    if ((threadIdx.x & 31) == 0) sm[threadIdx.x >> 5] = s;
    __syncthreads();
    if (threadIdx.x == 0) {
        float tot = 0.0f;
        for (int j = 0; j < 8; j++) tot += sm[j];
        cz[b] = SS - (int)(tot + 0.5f);
    }
}

// ---------------------------------------------------------------------------
// Kernel 4: LayerNorm over D=256, scatter to forward half and (involutive)
// reversed backward half of z[B,S,512].
// ---------------------------------------------------------------------------
__global__ void layernorm_rev(const float* __restrict__ x,
                              const float* __restrict__ gamma, const float* __restrict__ beta,
                              const int* __restrict__ cz, float* __restrict__ z) {
    int bs = blockIdx.x;                                  // 0..B*S-1
    int b = bs >> 11, s = bs & 2047;
    int d = threadIdx.x;                                  // 256 threads
    float v = x[(size_t)bs * DD + d];
    float sum = v, sq = v * v;
    for (int off = 16; off; off >>= 1) { sum += __shfl_xor(sum, off); sq += __shfl_xor(sq, off); }
    __shared__ float sm1[8], sm2[8], mv[2];
    if ((d & 31) == 0) { sm1[d >> 5] = sum; sm2[d >> 5] = sq; }
    __syncthreads();
    if (d == 0) {
        float ts = 0.0f, tq = 0.0f;
        for (int j = 0; j < 8; j++) { ts += sm1[j]; tq += sm2[j]; }
        float mean = ts * (1.0f / 256.0f);
        float var  = tq * (1.0f / 256.0f) - mean * mean;
        mv[0] = mean; mv[1] = rsqrtf(var + 1e-5f);
    }
    __syncthreads();
    float st = (v - mv[0]) * mv[1] * gamma[d] + beta[d];
    z[(size_t)bs * HH + d] = st;
    unsigned jrev = (unsigned)(2047 - s - cz[b]) & 2047u;
    z[((size_t)(b * SS) + jrev) * HH + DD + d] = st;
}

// ---------------------------------------------------------------------------
// Kernel 5: S4D diagonal recurrence, one wave32 per (b,h). Lane owns states
// n=lane and n=lane+32. Output: gelu(y + D*z), bf16, with the backward half
// scattered back through the reversal involution -> z2[M=B*S, K=512].
// ---------------------------------------------------------------------------
__global__ void s4d_scan(const float* __restrict__ z, const float4* __restrict__ prm,
                         const float* __restrict__ Dskip, const int* __restrict__ cz,
                         unsigned short* __restrict__ z2) {
    int wave = blockIdx.x * (blockDim.x >> 5) + (threadIdx.x >> 5);
    int lane = threadIdx.x & 31;
    int b = wave >> 9;          // H=512
    int h = wave & 511;
    float4 p0 = prm[h * NSS + lane];
    float4 p1 = prm[h * NSS + 32 + lane];
    float dsk = Dskip[h];
    int czb = cz[b];
    const float* zp = z + (size_t)(b * SS) * HH + h;
    float x0r = 0.0f, x0i = 0.0f, x1r = 0.0f, x1i = 0.0f;
    for (int s = 0; s < SS; ++s) {
        float zv = zp[(size_t)s * HH];
        float n0r = p0.x * x0r - p0.y * x0i + zv;
        float n0i = p0.x * x0i + p0.y * x0r;
        float n1r = p1.x * x1r - p1.y * x1i + zv;
        float n1i = p1.x * x1i + p1.y * x1r;
        x0r = n0r; x0i = n0i; x1r = n1r; x1i = n1i;
        float part = p0.z * x0r - p0.w * x0i + p1.z * x1r - p1.w * x1i;
        for (int off = 16; off; off >>= 1) part += __shfl_xor(part, off);
        float g = gelu_tanh(part + dsk * zv);
        if (lane == 0) {
            int sd = (h < DD) ? s : (int)((unsigned)(2047 - s - czb) & 2047u);
            z2[((size_t)(b * SS) + sd) * HH + h] = f2bf(g);
        }
    }
}

// ---------------------------------------------------------------------------
// Async global->LDS staging of one 32KB K-slab of the swizzled W matrix.
// Each of 256 threads copies 8 x 16B (GLOBAL_LOAD_ASYNC_TO_LDS_B128, GVS
// mode: saddr base + 32-bit vaddr offset). Tracked by ASYNCcnt.
// ---------------------------------------------------------------------------
__device__ __forceinline__ void stage_slab(unsigned long long wbase,
                                           unsigned short* dst, int kb, int tid) {
#pragma unroll
    for (int u = 0; u < 8; ++u) {
        unsigned ldsa = (unsigned)(uintptr_t)(dst + (u * 256 + tid) * 8);
        unsigned goff = (unsigned)(kb * 32768 + (u * 256 + tid) * 16);
        asm volatile("global_load_async_to_lds_b128 %0, %1, %2"
                     :: "v"(ldsa), "v"(goff), "s"(wbase) : "memory");
    }
}

// ---------------------------------------------------------------------------
// Kernel 6: fused GLU GEMM with bf16 WMMA. Workgroup = 8 waves covering
// 2 M-tiles x 4 N-tiles; B operand (W) is staged per-32K-slab into LDS with
// double-buffered async copies, shared by all 8 waves. Each wave computes a
// 16x64 'a' tile AND the matching 16x64 gate 'b' tile (8 f32 accumulators),
// then writes out = a*sigmoid(b) + residual.
// ---------------------------------------------------------------------------
__global__ void glu_gemm(const unsigned short* __restrict__ z2,
                         const unsigned short* __restrict__ Wsw,
                         const float* __restrict__ bglu,
                         const float* __restrict__ xin, float* __restrict__ out) {
    __shared__ unsigned short ldsW[2][16384];            // 2 x 32KB double buffer

    int tid   = threadIdx.x;
    int wv    = tid >> 5;                  // 0..7
    int lane  = tid & 31;
    int mtile = blockIdx.x * 2 + (wv >> 2);
    int ntile = wv & 3;
    int m0 = mtile * 16;
    int n0 = ntile * 64;
    int half = lane >> 4;
    int l15  = lane & 15;
    unsigned long long wbase = (unsigned long long)(uintptr_t)Wsw;

    v8f ca[4], cb[4];
#pragma unroll
    for (int j = 0; j < 4; j++) {
        float ba = bglu[n0 + j * 16 + l15];
        float bb = bglu[DD + n0 + j * 16 + l15];
#pragma unroll
        for (int r = 0; r < 8; r++) { ca[j][r] = ba; cb[j][r] = bb; }
    }

    // Prologue: stage slab 0
    stage_slab(wbase, &ldsW[0][0], 0, tid);
    asm volatile("s_wait_asynccnt 0x0" ::: "memory");
    __syncthreads();

    const unsigned short* arow = z2 + (size_t)(m0 + l15) * HH;
    for (int kb = 0; kb < 16; ++kb) {
        int cur = kb & 1;
        if (kb < 15) stage_slab(wbase, &ldsW[cur ^ 1][0], kb + 1, tid);

        // A tile (16x32 bf16): lane half 0 -> K {0..7,16..23}; half 1 -> K {8..15,24..31}
        const unsigned short* ap = arow + kb * 32 + half * 8;
        BfPack A;
        A.u[0] = *(const u32x4*)(ap);
        A.u[1] = *(const u32x4*)(ap + 16);
        if (kb < 15) __builtin_prefetch(arow + (kb + 1) * 32 + half * 8, 0, 0);

        const unsigned short* lw = &ldsW[cur][0];
#pragma unroll
        for (int j = 0; j < 4; j++) {
            int nbA = ntile * 4 + j;        // columns n0..n0+63      ('a' half)
            int nbB = 16 + ntile * 4 + j;   // columns 256+n0..       ('b' half)
            const unsigned short* bpa = lw + (nbA * 32 + lane) * 16;
            const unsigned short* bpb = lw + (nbB * 32 + lane) * 16;
            BfPack Ba, Bb;
            Ba.u[0] = *(const u32x4*)(bpa);
            Ba.u[1] = *(const u32x4*)(bpa + 8);
            Bb.u[0] = *(const u32x4*)(bpb);
            Bb.u[1] = *(const u32x4*)(bpb + 8);
            ca[j] = __builtin_amdgcn_wmma_f32_16x16x32_bf16(false, A.v, false, Ba.v,
                                                            (short)0, ca[j], false, false);
            cb[j] = __builtin_amdgcn_wmma_f32_16x16x32_bf16(false, A.v, false, Bb.v,
                                                            (short)0, cb[j], false, false);
        }

        if (kb < 15) {
            asm volatile("s_wait_asynccnt 0x0" ::: "memory");
            __syncthreads();
        }
    }

    // Epilogue: out = a * sigmoid(b) + residual. C layout: row = r + 8*half, col = lane%16.
#pragma unroll
    for (int j = 0; j < 4; j++) {
#pragma unroll
        for (int r = 0; r < 8; r++) {
            int row = m0 + r + 8 * half;
            int col = n0 + j * 16 + l15;
            float av = ca[j][r];
            float bv = cb[j][r];
            float sig = 1.0f / (1.0f + expf(-bv));
            size_t idx = (size_t)row * DD + col;
            out[idx] = av * sig + xin[idx];
        }
    }
}

// ---------------------------------------------------------------------------
extern "C" void kernel_launch(void* const* d_in, const int* in_sizes, int n_in,
                              void* d_out, int out_size, void* d_ws, size_t ws_size,
                              hipStream_t stream) {
    (void)in_sizes; (void)n_in; (void)out_size; (void)ws_size;
    const float* x      = (const float*)d_in[0];
    const float* mask   = (const float*)d_in[1];
    const float* log_dt = (const float*)d_in[2];
    const float* logAr  = (const float*)d_in[3];
    const float* Aim    = (const float*)d_in[4];
    const float* Bre    = (const float*)d_in[5];
    const float* Bim    = (const float*)d_in[6];
    const float* Cre    = (const float*)d_in[7];
    const float* Cim    = (const float*)d_in[8];
    const float* Dsk    = (const float*)d_in[9];
    const float* Wglu   = (const float*)d_in[10];
    const float* bglu   = (const float*)d_in[11];
    const float* gam    = (const float*)d_in[12];
    const float* bet    = (const float*)d_in[13];
    float* out = (float*)d_out;

    char* ws = (char*)d_ws;
    float*          z   = (float*)(ws);                               // 64 MiB
    unsigned short* z2  = (unsigned short*)(ws + 67108864ull);        // 32 MiB
    unsigned short* Wsw = (unsigned short*)(ws + 100663296ull);       // 512 KiB
    float4*         prm = (float4*)(ws + 101187584ull);               // 512 KiB
    int*            cz  = (int*)(ws + 101711872ull);                  // 64 B

    prep_params  <<<128,   256, 0, stream>>>(log_dt, logAr, Aim, Bre, Bim, Cre, Cim, prm);
    prep_w       <<<1024,  256, 0, stream>>>(Wglu, Wsw);
    count_zeros_k<<<16,    256, 0, stream>>>(mask, cz);
    layernorm_rev<<<32768, 256, 0, stream>>>(x, gam, bet, cz, z);
    s4d_scan     <<<1024,  256, 0, stream>>>(z, prm, Dsk, cz, z2);
    glu_gemm     <<<1024,  256, 0, stream>>>(z2, Wsw, bglu, x, out);
}
